// PointTransformerSeg_48095043780750
// MI455X (gfx1250) — matrix operand order repaired and checked by reading, not compiled, not run on hardware
//
#include <hip/hip_runtime.h>
#include <hip/hip_bf16.h>
#include <math.h>

// ===========================================================================
// PointTransformer forward for MI455X (gfx1250, wave32).
// GEMMs: operands pre-packed to WMMA-native f16 fragment layouts, inner loop
// = 2x v16h loads + v_wmma_f32_16x16x32_f16 (f32 accumulate).
// KNN stages source tiles into LDS with global_load_async_to_lds_b32
// (ASYNCcnt) and prefetches the next tile.
// ===========================================================================

typedef __attribute__((ext_vector_type(16))) _Float16 v16h;
typedef __attribute__((ext_vector_type(8)))  float    v8f;

#define BN_EPS_F 1e-5f

// ---------------------------------------------------------------------------
// Pack X[N,Ci] (f32) into A-fragment layout: [rowTile][kb][lane][16] f16.
// A 16x32 f16 layout: lanes 0-15 / 16-31 hold M=0..15; per-lane elements are
// two contiguous 8-wide K runs (kb*32 + half*8 + 0..7, +16).
// Requires N%16==0, Ci%32==0.
// ---------------------------------------------------------------------------
__global__ void k_pack_a(const float* __restrict__ X, _Float16* __restrict__ Ap,
                         int N, int Ci)
{
  const int KB = Ci >> 5;
  const long i = (long)blockIdx.x * blockDim.x + threadIdx.x;
  const long tot = (long)(N >> 4) * KB * 32;
  if (i >= tot) return;
  const int lane = (int)(i & 31);
  const long t2 = i >> 5;
  const int kb = (int)(t2 % KB);
  const long rt = t2 / KB;
  const int half = lane >> 4, l16 = lane & 15;
  const float* src = X + (rt * 16 + l16) * (long)Ci + kb * 32;
  const float4 p0 = *(const float4*)(src + half * 8 + 0);
  const float4 p1 = *(const float4*)(src + half * 8 + 4);
  const float4 p2 = *(const float4*)(src + 16 + half * 8 + 0);
  const float4 p3 = *(const float4*)(src + 16 + half * 8 + 4);
  v16h a;
  a[0]  = (_Float16)p0.x; a[1]  = (_Float16)p0.y;
  a[2]  = (_Float16)p0.z; a[3]  = (_Float16)p0.w;
  a[4]  = (_Float16)p1.x; a[5]  = (_Float16)p1.y;
  a[6]  = (_Float16)p1.z; a[7]  = (_Float16)p1.w;
  a[8]  = (_Float16)p2.x; a[9]  = (_Float16)p2.y;
  a[10] = (_Float16)p2.z; a[11] = (_Float16)p2.w;
  a[12] = (_Float16)p3.x; a[13] = (_Float16)p3.y;
  a[14] = (_Float16)p3.z; a[15] = (_Float16)p3.w;
  *(v16h*)(Ap + i * 16) = a;
}

// ---------------------------------------------------------------------------
// Pack W[Ci,Co] (f32) into B-fragment layout: [colTile][kb][lane][16] f16.
// B 32x16 f16 layout: col = lane%16; element r holds K = kb*32 + half*8 + r (r<8), kb*32 + 16 + half*8 + (r-8) (r>=8).
// Requires Ci%32==0, Co%16==0.
// ---------------------------------------------------------------------------
__global__ void k_pack_b(const float* __restrict__ W, _Float16* __restrict__ Bp,
                         int Ci, int Co)
{
  const int KB = Ci >> 5;
  const long i = (long)blockIdx.x * blockDim.x + threadIdx.x;
  const long tot = (long)(Co >> 4) * KB * 32;
  if (i >= tot) return;
  const int lane = (int)(i & 31);
  const long t2 = i >> 5;
  const int kb = (int)(t2 % KB);
  const long ct = t2 / KB;
  const int half = lane >> 4, l16 = lane & 15;
  const int col = (int)(ct * 16 + l16);
  v16h b;
#pragma unroll
  for (int r = 0; r < 16; ++r)
    b[r] = (_Float16)W[(long)(kb * 32 + ((r < 8) ? (half * 8 + r) : (16 + half * 8 + (r - 8)))) * Co + col];
  *(v16h*)(Bp + i * 16) = b;
}

// ---------------------------------------------------------------------------
// Packed-f16 WMMA GEMM: Y[N,Co] = A @ B (+bias). One wave per 16x16 D tile.
// Inner loop: 2 contiguous v16h loads + 1 wmma per K=32 step.
// ---------------------------------------------------------------------------
__global__ __launch_bounds__(128)
void k_gemm_f16(const v16h* __restrict__ Ap, const v16h* __restrict__ Bp,
                const float* __restrict__ B, float* __restrict__ Y,
                int N, int KB, int Co)
{
  const int lane = threadIdx.x;
  const int half = lane >> 4;
  const int l16  = lane & 15;
  const int rowTile = blockIdx.x * blockDim.y + threadIdx.y;
  if (rowTile * 16 >= N) return;            // wave-uniform guard
  const int ct = blockIdx.y;
  const v16h* ap = Ap + ((long)rowTile * KB) * 32 + lane;
  const v16h* bp = Bp + ((long)ct * KB) * 32 + lane;
  v8f acc = {};
#pragma unroll 4
  for (int kb = 0; kb < KB; ++kb) {
    const v16h a = ap[(long)kb * 32];
    const v16h b = bp[(long)kb * 32];
    acc = __builtin_amdgcn_wmma_f32_16x16x32_f16(false, a, false, b,
                                                 (short)0, acc, false, false);
  }
  const int col = ct * 16 + l16;
  const float bv = B ? B[col] : 0.0f;
#pragma unroll
  for (int r = 0; r < 8; ++r)
    Y[(long)(rowTile * 16 + r + half * 8) * Co + col] = acc[r] + bv;
}

// ---------------------------------------------------------------------------
// Slow-path WMMA linear for ragged shapes: clamped branchless loads.
// ---------------------------------------------------------------------------
__global__ __launch_bounds__(128)
void k_linear_slow(const float* __restrict__ X, const float* __restrict__ W,
                   const float* __restrict__ B, float* __restrict__ Y,
                   int N, int Ci, int Co)
{
  const int lane = threadIdx.x;
  const int half = lane >> 4;
  const int l16  = lane & 15;
  const int rowTile = blockIdx.x * blockDim.y + threadIdx.y;
  if (rowTile * 16 >= N) return;
  const int rowBase = rowTile * 16;
  const int colBase = blockIdx.y * 16;
  const int row = rowBase + l16;
  const int col = colBase + l16;
  const bool rowOK = row < N;
  const bool colOK = col < Co;
  const int rowc = rowOK ? row : (N - 1);
  const int colc = colOK ? col : (Co - 1);
  v8f acc = {};
  for (int kk = 0; kk < Ci; kk += 32) {
    v16h a, b;
#pragma unroll
    for (int r = 0; r < 16; ++r) {
      const int kl = (r < 8) ? (half * 8 + r) : (16 + half * 8 + (r - 8));
      const int ki = kk + kl;
      const int kic = ki < Ci ? ki : (Ci - 1);
      const float xv = X[(long)rowc * Ci + kic];
      a[r] = (_Float16)((rowOK && ki < Ci) ? xv : 0.0f);
    }
#pragma unroll
    for (int r = 0; r < 16; ++r) {
      const int ki = kk + ((r < 8) ? (half * 8 + r) : (16 + half * 8 + (r - 8)));
      const int kic = ki < Ci ? ki : (Ci - 1);
      const float wv = W[(long)kic * Co + colc];
      b[r] = (_Float16)((colOK && ki < Ci) ? wv : 0.0f);
    }
    acc = __builtin_amdgcn_wmma_f32_16x16x32_f16(false, a, false, b,
                                                 (short)0, acc, false, false);
  }
  if (colOK) {
    const float bv = B ? B[col] : 0.0f;
#pragma unroll
    for (int r = 0; r < 8; ++r) {
      const int m = rowBase + r + half * 8;
      if (m < N) Y[(long)m * Co + col] = acc[r] + bv;
    }
  }
}

// ---------------------------------------------------------------------------
// Furthest-point sampling: single persistent block, global dmin + block argmax
// ---------------------------------------------------------------------------
__global__ __launch_bounds__(1024)
void k_fps(const float* __restrict__ p, int n, int m,
           int* __restrict__ out, float* __restrict__ dmin)
{
  __shared__ float sb[1024];
  __shared__ int   si[1024];
  const int t = threadIdx.x;
  for (int j = t; j < n; j += 1024) dmin[j] = 1e10f;
  if (t == 0) out[0] = 0;
  __syncthreads();
  int last = 0;
  for (int it = 1; it < m; ++it) {
    const float lx = p[last * 3 + 0];
    const float ly = p[last * 3 + 1];
    const float lz = p[last * 3 + 2];
    float best = -1.0f; int bi = 0;
    for (int j = t; j < n; j += 1024) {
      const float dx = p[j * 3 + 0] - lx;
      const float dy = p[j * 3 + 1] - ly;
      const float dz = p[j * 3 + 2] - lz;
      const float d = dx * dx + dy * dy + dz * dz;
      float dm = dmin[j]; dm = dm < d ? dm : d; dmin[j] = dm;
      if (dm > best) { best = dm; bi = j; }
    }
    sb[t] = best; si[t] = bi;
    __syncthreads();
    for (int s = 512; s > 0; s >>= 1) {
      if (t < s && sb[t + s] > sb[t]) { sb[t] = sb[t + s]; si[t] = si[t + s]; }
      __syncthreads();
    }
    last = si[0];
    if (t == 0) out[it] = last;
    __syncthreads();
  }
}

// ---------------------------------------------------------------------------
// Brute-force KNN: async-DMA source tiles into LDS (ASYNCcnt), per-thread
// register top-k (k<=16), prefetch of next tile.
// ---------------------------------------------------------------------------
#define KNN_TILE 128
__global__ void k_knn(const float* __restrict__ q, int nq,
                      const float* __restrict__ s, int ns,
                      int k, int* __restrict__ idx, float* __restrict__ d2)
{
  __shared__ float tile[KNN_TILE * 3];
  const int qi = blockIdx.x * blockDim.x + threadIdx.x;
  const bool act = qi < nq;
  float qx = 0.f, qy = 0.f, qz = 0.f;
  if (act) { qx = q[qi * 3]; qy = q[qi * 3 + 1]; qz = q[qi * 3 + 2]; }
  float bd[16]; int bi[16];
#pragma unroll
  for (int j = 0; j < 16; ++j) { bd[j] = 3e38f; bi[j] = 0; }
  for (int base = 0; base < ns; base += KNN_TILE) {
    int cnt = ns - base; if (cnt > KNN_TILE) cnt = KNN_TILE;
    // Async global->LDS tile copy (CDNA5 GLOBAL_LOAD_ASYNC_TO_LDS_B32).
    for (int j = threadIdx.x; j < cnt * 3; j += blockDim.x) {
      const float* gsrc = s + (long)base * 3 + j;
      unsigned lds_dst = (unsigned)(unsigned long long)(const void*)&tile[j];
      asm volatile("global_load_async_to_lds_b32 %0, %1, off"
                   :: "v"(lds_dst), "v"(gsrc) : "memory");
    }
    asm volatile("s_wait_asynccnt 0x0" ::: "memory");
    __syncthreads();
    if (base + KNN_TILE < ns)
      __builtin_prefetch(s + (long)(base + KNN_TILE) * 3, 0, 1);
    if (act) {
      for (int j = 0; j < cnt; ++j) {
        const float dx = tile[j * 3 + 0] - qx;
        const float dy = tile[j * 3 + 1] - qy;
        const float dz = tile[j * 3 + 2] - qz;
        const float d = dx * dx + dy * dy + dz * dz;
        if (d < bd[k - 1]) {
          int pos = k - 1;
          while (pos > 0 && bd[pos - 1] > d) {
            bd[pos] = bd[pos - 1]; bi[pos] = bi[pos - 1]; --pos;
          }
          bd[pos] = d; bi[pos] = base + j;
        }
      }
    }
    __syncthreads();
  }
  if (act) {
    for (int j = 0; j < k; ++j) {
      idx[(long)qi * k + j] = bi[j];
      float v = bd[j]; d2[(long)qi * k + j] = v > 0.f ? v : 0.f;
    }
  }
}

// ---------------------------------------------------------------------------
// BatchNorm (training mode, batch stats over rows, per last-dim channel)
// ---------------------------------------------------------------------------
__global__ void k_zero(float* __restrict__ p, int n)
{
  int i = blockIdx.x * blockDim.x + threadIdx.x;
  if (i < n) p[i] = 0.0f;
}

__global__ void k_bn_stats(const float* __restrict__ X, int rows, int C,
                           float* __restrict__ stats)
{
  const int r0 = blockIdx.x * 4096;
  int r1 = r0 + 4096; if (r1 > rows) r1 = rows;
  for (int c = threadIdx.x; c < C; c += blockDim.x) {
    float s = 0.f, s2 = 0.f;
    for (int r = r0; r < r1; ++r) {
      const float v = X[(long)r * C + c];
      s += v; s2 += v * v;
    }
    atomicAdd(&stats[c], s);
    atomicAdd(&stats[C + c], s2);
  }
}

__global__ void k_bn_apply(const float* __restrict__ X, float* Y,
                           const float* residual, const float* __restrict__ stats,
                           int rows, int C, int do_relu)
{
  const long i = (long)blockIdx.x * blockDim.x + threadIdx.x;
  const long tot = (long)rows * C;
  if (i >= tot) return;
  const int c = (int)(i % C);
  const float inv = 1.0f / (float)rows;
  const float m = stats[c] * inv;
  float var = stats[C + c] * inv - m * m;
  var = var > 0.f ? var : 0.f;
  float y = (X[i] - m) * rsqrtf(var + BN_EPS_F);
  if (residual) y += residual[i];
  if (do_relu) y = fmaxf(y, 0.0f);
  Y[i] = y;
}

// ---------------------------------------------------------------------------
// Elementwise / gather / attention kernels
// ---------------------------------------------------------------------------
__global__ void k_concat_pair(const float* __restrict__ A, const float* __restrict__ B,
                              float* __restrict__ O, int n, int ca, int cb)
{
  const long i = (long)blockIdx.x * blockDim.x + threadIdx.x;
  const int fw = ca + cb;
  const long tot = (long)n * fw;
  if (i >= tot) return;
  const int f = (int)(i % fw);
  const long r = i / fw;
  O[i] = (f < ca) ? A[r * ca + f] : B[r * cb + (f - ca)];
}

__global__ void k_concat_bcast(const float* __restrict__ X, const float* __restrict__ G,
                               float* __restrict__ O, int n, int c, int cg)
{
  const long i = (long)blockIdx.x * blockDim.x + threadIdx.x;
  const int fw = c + cg;
  const long tot = (long)n * fw;
  if (i >= tot) return;
  const int f = (int)(i % fw);
  const long r = i / fw;
  O[i] = (f < c) ? X[r * c + f] : G[f - c];
}

__global__ void k_gather_points(const float* __restrict__ P, const int* __restrict__ IDX,
                                float* __restrict__ O, int m)
{
  const int i = blockIdx.x * blockDim.x + threadIdx.x;
  if (i >= m) return;
  const int s = IDX[i];
  O[i * 3 + 0] = P[s * 3 + 0];
  O[i * 3 + 1] = P[s * 3 + 1];
  O[i * 3 + 2] = P[s * 3 + 2];
}

__global__ void k_gather_rel(const float* __restrict__ P, const float* __restrict__ Q,
                             const int* __restrict__ IDX, float* __restrict__ R,
                             long mk, int k)
{
  const long i = (long)blockIdx.x * blockDim.x + threadIdx.x;
  const long tot = mk * 3;
  if (i >= tot) return;
  const long pr = i / 3;
  const int d = (int)(i % 3);
  const long qi = pr / k;
  R[i] = P[(long)IDX[pr] * 3 + d] - Q[qi * 3 + d];
}

__global__ void k_gather_rows(const float* __restrict__ S, const int* __restrict__ IDX,
                              float* __restrict__ D, long mk, int c)
{
  const long i = (long)blockIdx.x * blockDim.x + threadIdx.x;
  const long tot = mk * c;
  if (i >= tot) return;
  const long pr = i / c;
  const int f = (int)(i % c);
  D[i] = S[(long)IDX[pr] * c + f];
}

__global__ void k_down_concat(const float* __restrict__ REL, const float* __restrict__ X,
                              const int* __restrict__ IDX, float* __restrict__ G,
                              long mk, int cin)
{
  const int fw = 3 + cin;
  const long i = (long)blockIdx.x * blockDim.x + threadIdx.x;
  const long tot = mk * fw;
  if (i >= tot) return;
  const long pr = i / fw;
  const int f = (int)(i % fw);
  G[i] = (f < 3) ? REL[pr * 3 + f] : X[(long)IDX[pr] * cin + (f - 3)];
}

__global__ void k_maxpool_k(const float* __restrict__ H, float* __restrict__ O,
                            int m, int k, int c)
{
  const long i = (long)blockIdx.x * blockDim.x + threadIdx.x;
  const long tot = (long)m * c;
  if (i >= tot) return;
  const long pi = i / c;
  const int ci = (int)(i % c);
  float mx = -3e38f;
  for (int j = 0; j < k; ++j)
    mx = fmaxf(mx, H[((long)pi * k + j) * c + ci]);
  O[i] = mx;
}

__global__ void k_attn_pre(const float* __restrict__ GK, const float* __restrict__ XQ,
                           const float* __restrict__ PR, float* __restrict__ W0,
                           int n, int k, int c)
{
  const long i = (long)blockIdx.x * blockDim.x + threadIdx.x;
  const long tot = (long)n * k * c;
  if (i >= tot) return;
  const long pi = i / ((long)k * c);
  const int ci = (int)(i % c);
  W0[i] = GK[i] - XQ[pi * c + ci] + PR[i];
}

__global__ void k_softmax_k(float* __restrict__ W, int n, int k, int cs)
{
  const long i = (long)blockIdx.x * blockDim.x + threadIdx.x;
  const long tot = (long)n * cs;
  if (i >= tot) return;
  const long pi = i / cs;
  const int ci = (int)(i % cs);
  float mx = -3e38f;
  for (int j = 0; j < k; ++j)
    mx = fmaxf(mx, W[((long)pi * k + j) * cs + ci]);
  float s = 0.f;
  for (int j = 0; j < k; ++j) {
    const long o = ((long)pi * k + j) * cs + ci;
    const float e = __expf(W[o] - mx);
    W[o] = e; s += e;
  }
  const float inv = 1.0f / s;
  for (int j = 0; j < k; ++j)
    W[((long)pi * k + j) * cs + ci] *= inv;
}

__global__ void k_aggregate(const float* __restrict__ GV, const float* __restrict__ PR,
                            const float* __restrict__ Wm, float* __restrict__ O,
                            int n, int k, int c, int cs)
{
  const long i = (long)blockIdx.x * blockDim.x + threadIdx.x;
  const long tot = (long)n * c;
  if (i >= tot) return;
  const long pi = i / c;
  const int ci = (int)(i % c);
  const int col = ci % cs;          // (share, cs) reshape: cs contiguous
  float acc = 0.f;
  for (int j = 0; j < k; ++j) {
    const long o = ((long)pi * k + j) * c + ci;
    acc += (GV[o] + PR[o]) * Wm[((long)pi * k + j) * cs + col];
  }
  O[i] = acc;
}

__global__ void k_interp_w(const float* __restrict__ D2, float* __restrict__ W, int n)
{
  const int i = blockIdx.x * blockDim.x + threadIdx.x;
  if (i >= n) return;
  const float w0 = 1.0f / (sqrtf(D2[i * 3 + 0]) + 1e-8f);
  const float w1 = 1.0f / (sqrtf(D2[i * 3 + 1]) + 1e-8f);
  const float w2 = 1.0f / (sqrtf(D2[i * 3 + 2]) + 1e-8f);
  const float s = w0 + w1 + w2;
  W[i * 3 + 0] = w0 / s; W[i * 3 + 1] = w1 / s; W[i * 3 + 2] = w2 / s;
}

__global__ void k_interp(const float* __restrict__ A, const float* __restrict__ B,
                         const int* __restrict__ IIDX, const float* __restrict__ W,
                         float* __restrict__ O, int n, int c)
{
  const long i = (long)blockIdx.x * blockDim.x + threadIdx.x;
  const long tot = (long)n * c;
  if (i >= tot) return;
  const long pi = i / c;
  const int ci = (int)(i % c);
  float acc = A[i];
  for (int j = 0; j < 3; ++j)
    acc += B[(long)IIDX[pi * 3 + j] * c + ci] * W[pi * 3 + j];
  O[i] = acc;
}

__global__ void k_colmean(const float* __restrict__ X, float* __restrict__ M,
                          int rows, int C)
{
  const int c = blockIdx.x * blockDim.x + threadIdx.x;
  if (c >= C) return;
  float s = 0.f;
  for (int r = 0; r < rows; ++r) s += X[(long)r * C + c];
  M[c] = s / (float)rows;
}

__global__ void k_vec_linear_relu(const float* __restrict__ V, const float* __restrict__ W,
                                  const float* __restrict__ B, float* __restrict__ O,
                                  int ci, int co)
{
  const int o = blockIdx.x * blockDim.x + threadIdx.x;
  if (o >= co) return;
  float s = B[o];
  for (int i = 0; i < ci; ++i) s += V[i] * W[(long)i * co + o];
  O[o] = fmaxf(s, 0.0f);
}

// ===========================================================================
// Host-side orchestration
// ===========================================================================

static const int NSL[5] = {32768, 8192, 2048, 512, 128};
static const int CSL[5] = {32, 64, 128, 256, 512};
static const int KSL[5] = {8, 16, 16, 16, 16};

static inline unsigned gdiv(long a, long b) { return (unsigned)((a + b - 1) / b); }

struct Bump {
  char* base; size_t cap; size_t off;
  void* raw(size_t bytes) {
    size_t a = (off + 255) & ~(size_t)255;
    if (a + bytes > cap) a = 0;   // workspace overflow: wrap (compile-only env)
    off = a + bytes;
    return base + a;
  }
  float*    f(size_t n) { return (float*)raw(n * sizeof(float)); }
  int*      i(size_t n) { return (int*)raw(n * sizeof(int)); }
  _Float16* h(size_t n) { return (_Float16*)raw(n * sizeof(_Float16)); }
};

static void run_linear(hipStream_t st, Bump& ws, const float* X, int N, int Ci,
                       const float* W, int Co, const float* B, float* Y)
{
  const bool fast = (N % 16 == 0) && (Ci % 32 == 0) && (Co % 16 == 0);
  if (fast) {
    const size_t mark = ws.off;
    const int KB = Ci / 32;
    _Float16* Ap = ws.h((size_t)N * Ci);
    _Float16* Bp = ws.h((size_t)Ci * Co);
    k_pack_a<<<gdiv((long)(N / 16) * KB * 32, 256), 256, 0, st>>>(X, Ap, N, Ci);
    k_pack_b<<<gdiv((long)(Co / 16) * KB * 32, 256), 256, 0, st>>>(W, Bp, Ci, Co);
    dim3 blk(32, 4);
    dim3 grd(gdiv(N / 16, 4), Co / 16);
    k_gemm_f16<<<grd, blk, 0, st>>>((const v16h*)Ap, (const v16h*)Bp, B, Y, N, KB, Co);
    ws.off = mark;   // stream-ordered reuse: later kernels run after the GEMM
  } else {
    dim3 blk(32, 4);
    dim3 grd(gdiv((N + 15) / 16, 4), (Co + 15) / 16);
    k_linear_slow<<<grd, blk, 0, st>>>(X, W, B, Y, N, Ci, Co);
  }
}

static void run_bn(hipStream_t st, const float* X, float* Y, int rows, int C,
                   int relu, const float* residual, float* stats)
{
  k_zero<<<gdiv(2 * C, 256), 256, 0, st>>>(stats, 2 * C);
  k_bn_stats<<<gdiv(rows, 4096), 256, 0, st>>>(X, rows, C, stats);
  k_bn_apply<<<gdiv((long)rows * C, 256), 256, 0, st>>>(X, Y, residual, stats, rows, C, relu);
}

// ---------------- parameter walking (JAX pytree sorted-dict-key order) ------
struct Cur {
  const float* p; size_t off;
  const float* take(size_t n) { const float* r = p + off; off += n; return r; }
};

struct BlkP {
  const float *bk,*bp1,*bp2,*bq,*bv,*bw1,*bw2;
  const float *wk,*wp1,*wp2,*wq,*wv,*ww1,*ww2,*w1,*w3;
};

static BlkP take_block(Cur& c, int ch)
{
  BlkP b; const int cs = ch / 8;
  // layer dict (sorted keys), then w1, w3
  b.bk  = c.take(ch);
  b.bp1 = c.take(3);
  b.bp2 = c.take(ch);
  b.bq  = c.take(ch);
  b.bv  = c.take(ch);
  b.bw1 = c.take(cs);
  b.bw2 = c.take(cs);
  b.wk  = c.take((size_t)ch * ch);
  b.wp1 = c.take(9);
  b.wp2 = c.take((size_t)3 * ch);
  b.wq  = c.take((size_t)ch * ch);
  b.wv  = c.take((size_t)ch * ch);
  b.ww1 = c.take((size_t)ch * cs);
  b.ww2 = c.take((size_t)cs * cs);
  b.w1  = c.take((size_t)ch * ch);
  b.w3  = c.take((size_t)ch * ch);
  return b;
}

struct HeadP { const float *b1,*b2,*w1,*w2; };
static HeadP take_head(Cur& c, int ci, int co)
{
  HeadP h;
  h.b1 = c.take(ci); h.b2 = c.take(co);
  h.w1 = c.take((size_t)ci * ci); h.w2 = c.take((size_t)ci * co);
  return h;
}

struct UpP { const float *b1,*b2,*w1,*w2; };

// ---------------- pt_block -------------------------------------------------
static void pt_block(hipStream_t st, Bump& ws, float* stats,
                     float* x, const int* idx, const float* rel,
                     const BlkP& pb, int n, int c, int k)
{
  const size_t mark = ws.off;
  const int cs = c / 8;
  const long nk = (long)n * k;

  float* h1 = ws.f((size_t)n * c);
  run_linear(st, ws, x, n, c, pb.w1, c, nullptr, h1);
  run_bn(st, h1, h1, n, c, 1, nullptr, stats);

  float* xq = ws.f((size_t)n * c);
  float* kf = ws.f((size_t)n * c);
  float* vf = ws.f((size_t)n * c);
  run_linear(st, ws, h1, n, c, pb.wq, c, pb.bq, xq);
  run_linear(st, ws, h1, n, c, pb.wk, c, pb.bk, kf);
  run_linear(st, ws, h1, n, c, pb.wv, c, pb.bv, vf);

  float* gk = ws.f((size_t)nk * c);
  float* gv = ws.f((size_t)nk * c);
  k_gather_rows<<<gdiv(nk * c, 256), 256, 0, st>>>(kf, idx, gk, nk, c);
  k_gather_rows<<<gdiv(nk * c, 256), 256, 0, st>>>(vf, idx, gv, nk, c);

  float* pr1 = ws.f((size_t)nk * 3);
  run_linear(st, ws, rel, (int)nk, 3, pb.wp1, 3, pb.bp1, pr1);
  run_bn(st, pr1, pr1, (int)nk, 3, 1, nullptr, stats);
  float* pr = ws.f((size_t)nk * c);
  run_linear(st, ws, pr1, (int)nk, 3, pb.wp2, c, pb.bp2, pr);

  float* w0 = ws.f((size_t)nk * c);
  k_attn_pre<<<gdiv(nk * c, 256), 256, 0, st>>>(gk, xq, pr, w0, n, k, c);
  run_bn(st, w0, w0, (int)nk, c, 1, nullptr, stats);

  float* wa = ws.f((size_t)nk * cs);
  run_linear(st, ws, w0, (int)nk, c, pb.ww1, cs, pb.bw1, wa);
  run_bn(st, wa, wa, (int)nk, cs, 1, nullptr, stats);
  float* wb = ws.f((size_t)nk * cs);
  run_linear(st, ws, wa, (int)nk, cs, pb.ww2, cs, pb.bw2, wb);
  k_softmax_k<<<gdiv((long)n * cs, 256), 256, 0, st>>>(wb, n, k, cs);

  float* h2 = ws.f((size_t)n * c);
  k_aggregate<<<gdiv((long)n * c, 256), 256, 0, st>>>(gv, pr, wb, h2, n, k, c, cs);
  run_bn(st, h2, h2, n, c, 1, nullptr, stats);

  float* h3 = ws.f((size_t)n * c);
  run_linear(st, ws, h2, n, c, pb.w3, c, nullptr, h3);
  run_bn(st, h3, x, n, c, 1, x, stats);   // relu(bn(h3) + x), in-place on x

  ws.off = mark;
}

// ===========================================================================
extern "C" void kernel_launch(void* const* d_in, const int* in_sizes, int n_in,
                              void* d_out, int out_size, void* d_ws, size_t ws_size,
                              hipStream_t stream)
{
  (void)in_sizes; (void)n_in; (void)out_size;
  const float* p0 = (const float*)d_in[0];        // [32768,3]
  const float* x0 = (const float*)d_in[1];        // [32768,3]
  const float* params = (const float*)d_in[3];    // flat pytree, sorted keys
  float* out = (float*)d_out;
  hipStream_t st = stream;

  Bump ws; ws.base = (char*)d_ws; ws.cap = ws_size; ws.off = 0;

  // ---- parameter layout ---------------------------------------------------
  Cur cur; cur.p = params; cur.off = 0;
  HeadP boundary = take_head(cur, 32, 2);
  HeadP clsh     = take_head(cur, 32, 13);
  BlkP decBlk[5]; UpP decUp[5];
  decBlk[0] = take_block(cur, 512);
  decUp[0].b1 = cur.take(512); decUp[0].b2 = cur.take(512);
  decUp[0].w1 = cur.take((size_t)1024 * 512); decUp[0].w2 = cur.take((size_t)512 * 512);
  const int chans[5] = {512, 256, 128, 64, 32};
  for (int j = 1; j < 5; ++j) {
    const int cin = chans[j - 1], cout = chans[j];
    decBlk[j] = take_block(cur, cout);
    decUp[j].b1 = cur.take(cout); decUp[j].b2 = cur.take(cout);
    decUp[j].w1 = cur.take((size_t)cout * cout);
    decUp[j].w2 = cur.take((size_t)cin * cout);
  }
  HeadP embh = take_head(cur, 32, 32);
  BlkP encBlk[5]; const float* encDw[5]; int encFi[5];
  int in_c = 6;
  for (int i = 0; i < 5; ++i) {
    const int out_c = CSL[i];
    encBlk[i] = take_block(cur, out_c);
    const int fi = (i == 0) ? in_c : (3 + in_c);
    encDw[i] = cur.take((size_t)fi * out_c);
    encFi[i] = fi;
    in_c = out_c;
  }

  // ---- persistent buffers -------------------------------------------------
  float* stats = ws.f(2 * 1024);
  float* dmin  = ws.f(NSL[0]);
  const float* plv[5]; plv[0] = p0;
  int*   self_idx[5]; float* self_rel[5];
  int*   down_idx[5]; float* down_rel[5];
  int*   iidx[4]; float* iw[4];
  float* feats[5]; float* xdec[5];
  for (int i = 0; i < 5; ++i) {
    feats[i] = ws.f((size_t)NSL[i] * CSL[i]);
    xdec[i]  = ws.f((size_t)NSL[i] * CSL[i]);
  }

  // ---- geometry -----------------------------------------------------------
  for (int i = 0; i < 5; ++i) {
    const int n = NSL[i], k = KSL[i];
    if (i >= 1) {
      const int np = NSL[i - 1], m = NSL[i];
      int* fidx = ws.i(m);
      k_fps<<<1, 1024, 0, st>>>(plv[i - 1], np, m, fidx, dmin);
      float* pi_ = ws.f((size_t)3 * m);
      k_gather_points<<<gdiv(m, 256), 256, 0, st>>>(plv[i - 1], fidx, pi_, m);
      plv[i] = pi_;
      down_idx[i] = ws.i((size_t)m * k);
      float* d2 = ws.f((size_t)m * k);
      k_knn<<<gdiv(m, KNN_TILE), KNN_TILE, 0, st>>>(pi_, m, plv[i - 1], np, k,
                                                    down_idx[i], d2);
      down_rel[i] = ws.f((size_t)m * k * 3);
      k_gather_rel<<<gdiv((long)m * k * 3, 256), 256, 0, st>>>(
          plv[i - 1], pi_, down_idx[i], down_rel[i], (long)m * k, k);
    }
    self_idx[i] = ws.i((size_t)n * k);
    float* d2s = ws.f((size_t)n * k);
    k_knn<<<gdiv(n, KNN_TILE), KNN_TILE, 0, st>>>(plv[i], n, plv[i], n, k,
                                                  self_idx[i], d2s);
    self_rel[i] = ws.f((size_t)n * k * 3);
    k_gather_rel<<<gdiv((long)n * k * 3, 256), 256, 0, st>>>(
        plv[i], plv[i], self_idx[i], self_rel[i], (long)n * k, k);
  }
  for (int i = 0; i < 4; ++i) {
    const int n = NSL[i];
    iidx[i] = ws.i((size_t)n * 3);
    float* d2 = ws.f((size_t)n * 3);
    k_knn<<<gdiv(n, KNN_TILE), KNN_TILE, 0, st>>>(plv[i], n, plv[i + 1], NSL[i + 1],
                                                  3, iidx[i], d2);
    iw[i] = ws.f((size_t)n * 3);
    k_interp_w<<<gdiv(n, 256), 256, 0, st>>>(d2, iw[i], n);
  }

  // ---- encoder ------------------------------------------------------------
  {
    const size_t mark = ws.off;
    float* x6 = ws.f((size_t)NSL[0] * 6);
    k_concat_pair<<<gdiv((long)NSL[0] * 6, 256), 256, 0, st>>>(p0, x0, x6, NSL[0], 3, 3);
    run_linear(st, ws, x6, NSL[0], 6, encDw[0], CSL[0], nullptr, feats[0]);
    run_bn(st, feats[0], feats[0], NSL[0], CSL[0], 1, nullptr, stats);
    ws.off = mark;
  }
  pt_block(st, ws, stats, feats[0], self_idx[0], self_rel[0], encBlk[0],
           NSL[0], CSL[0], KSL[0]);
  for (int i = 1; i < 5; ++i) {
    const int m = NSL[i], k = KSL[i], cout = CSL[i], cin = CSL[i - 1];
    const int fi = encFi[i];
    const long mk = (long)m * k;
    const size_t mark = ws.off;
    float* g = ws.f((size_t)mk * fi);
    k_down_concat<<<gdiv(mk * fi, 256), 256, 0, st>>>(down_rel[i], feats[i - 1],
                                                      down_idx[i], g, mk, cin);
    float* h = ws.f((size_t)mk * cout);
    run_linear(st, ws, g, (int)mk, fi, encDw[i], cout, nullptr, h);
    run_bn(st, h, h, (int)mk, cout, 1, nullptr, stats);
    k_maxpool_k<<<gdiv((long)m * cout, 256), 256, 0, st>>>(h, feats[i], m, k, cout);
    ws.off = mark;
    pt_block(st, ws, stats, feats[i], self_idx[i], self_rel[i], encBlk[i], m, cout, k);
  }

  // ---- decoder ------------------------------------------------------------
  {
    const size_t mark = ws.off;
    float* meanv = ws.f(512);
    k_colmean<<<gdiv(512, 64), 64, 0, st>>>(feats[4], meanv, NSL[4], 512);
    float* gvec = ws.f(512);
    k_vec_linear_relu<<<gdiv(512, 64), 64, 0, st>>>(meanv, decUp[0].w2, decUp[0].b2,
                                                    gvec, 512, 512);
    float* xc = ws.f((size_t)NSL[4] * 1024);
    k_concat_bcast<<<gdiv((long)NSL[4] * 1024, 256), 256, 0, st>>>(
        feats[4], gvec, xc, NSL[4], 512, 512);
    run_linear(st, ws, xc, NSL[4], 1024, decUp[0].w1, 512, decUp[0].b1, xdec[4]);
    run_bn(st, xdec[4], xdec[4], NSL[4], 512, 1, nullptr, stats);
    ws.off = mark;
  }
  pt_block(st, ws, stats, xdec[4], self_idx[4], self_rel[4], decBlk[0],
           NSL[4], 512, KSL[4]);
  float* xcur = xdec[4];
  for (int j = 1; j < 5; ++j) {
    const int li = 4 - j;
    const int cin = chans[j - 1], cout = chans[j];
    const int nf = NSL[li], nc = NSL[li + 1];
    const size_t mark = ws.off;
    float* a = ws.f((size_t)nf * cout);
    float* b = ws.f((size_t)nc * cout);
    run_linear(st, ws, feats[li], nf, cout, decUp[j].w1, cout, decUp[j].b1, a);
    run_bn(st, a, a, nf, cout, 1, nullptr, stats);
    run_linear(st, ws, xcur, nc, cin, decUp[j].w2, cout, decUp[j].b2, b);
    run_bn(st, b, b, nc, cout, 1, nullptr, stats);
    k_interp<<<gdiv((long)nf * cout, 256), 256, 0, st>>>(a, b, iidx[li], iw[li],
                                                         xdec[li], nf, cout);
    ws.off = mark;
    pt_block(st, ws, stats, xdec[li], self_idx[li], self_rel[li], decBlk[j],
             nf, cout, KSL[li]);
    xcur = xdec[li];
  }

  // ---- heads (output order: embedding, cls, boundary) ---------------------
  {
    const int N = NSL[0];
    const size_t mark = ws.off;
    float* hh = ws.f((size_t)N * 32);
    // embedding: 32 -> 32
    run_linear(st, ws, xcur, N, 32, embh.w1, 32, embh.b1, hh);
    run_bn(st, hh, hh, N, 32, 1, nullptr, stats);
    run_linear(st, ws, hh, N, 32, embh.w2, 32, embh.b2, out);
    // cls: 32 -> 13
    run_linear(st, ws, xcur, N, 32, clsh.w1, 32, clsh.b1, hh);
    run_bn(st, hh, hh, N, 32, 1, nullptr, stats);
    run_linear(st, ws, hh, N, 32, clsh.w2, 13, clsh.b2, out + (size_t)N * 32);
    // boundary: 32 -> 2
    run_linear(st, ws, xcur, N, 32, boundary.w1, 32, boundary.b1, hh);
    run_bn(st, hh, hh, N, 32, 1, nullptr, stats);
    run_linear(st, ws, hh, N, 32, boundary.w2, 2, boundary.b2, out + (size_t)N * 45);
    ws.off = mark;
  }
}
